// LabelwiseAttention_53781580481065
// MI455X (gfx1250) — matrix operand order repaired and checked by reading, not compiled, not run on hardware
//
#include <hip/hip_runtime.h>

typedef __attribute__((ext_vector_type(8)))  float   v8f;
typedef __attribute__((ext_vector_type(16))) __bf16  v16bf;
typedef __attribute__((ext_vector_type(8)))  __bf16  v8bf;
typedef __attribute__((ext_vector_type(4)))  __bf16  v4bf;

#define B_    4
#define S_    4096
#define D_    256
#define C_    8921
#define CT    16               // classes per workgroup (one WMMA M tile)
#define ROWP  (S_ + 4)         // f32 row stride: 4100 dw == 4 (mod 64) -> conflict-free tiles
#define KC    32               // phase-3 K chunk staged in LDS per iteration
#define KP    (KC + 8)         // padded k stride (bf16): 80 B rows -> 16B-aligned v8bf reads
#define NKS   (S_ / KC)        // 128 staged chunks

__global__ __launch_bounds__(256)
void labelwise_attn_fused(const float* __restrict__ x,
                          const float* __restrict__ W,
                          float* __restrict__ logits,
                          float* __restrict__ attn)
{
    __shared__ float  s_scores[CT * ROWP];   // 262,400 B : f32 scores -> bf16 attention
    __shared__ __bf16 s_xs[2][D_ * KP];      //  40,960 B : double-buffered x^T chunks (bf16)
    __shared__ __bf16 s_w[CT * D_];          //   8,192 B : W tile (bf16)

    const int tid  = threadIdx.x;
    const int lane = tid & 31;
    const int wave = tid >> 5;
    const int hi   = lane >> 4;   // K-half select in this code's contiguous k = 16*hi + s order
    const int lo   = lane & 15;

    const int c0 = blockIdx.x * CT;
    const int b  = blockIdx.y;

    const float* __restrict__ xb = x + (size_t)b * S_ * D_;

    // ---------------- Phase 0: stage W tile as bf16 (A matrix of GEMM 1) ----------------
    for (int i = tid; i < CT * D_; i += 256) {
        const int r = i >> 8;          // / D_
        const int d = i & (D_ - 1);
        const int c = c0 + r;
        const float w = (c < C_) ? W[(size_t)c * D_ + d] : 0.0f;
        s_w[r * D_ + d] = (__bf16)w;
    }
    __syncthreads();

    // ---------------- Phase 1: scores[16][S] = Wt(16xD) * x^T(DxS) ----------------
    v16bf a1[8];
    #pragma unroll
    for (int kk = 0; kk < 8; ++kk) {
        const int kb = kk * 32;
        // lane<16: K = kb+0..15 ; lane>=16: K = kb+16..31
        const v8bf g1 = *(const v8bf*)&s_w[lo * D_ + kb + hi * 16];
        const v8bf g2 = *(const v8bf*)&s_w[lo * D_ + kb + hi * 16 + 8];
        v16bf a;
        #pragma unroll
        for (int j = 0; j < 8; ++j) { a[j] = g1[j]; a[8 + j] = g2[j]; }
        a1[kk] = a;
    }

    for (int t = wave; t < S_ / 16; t += 8) {        // 32 N-tiles per wave, uniform
        const int s0 = t * 16;
        const float* __restrict__ xrow = xb + (size_t)(s0 + lo) * D_ + hi * 16;
        v8f acc = {0.f, 0.f, 0.f, 0.f, 0.f, 0.f, 0.f, 0.f};
        #pragma unroll
        for (int kk = 0; kk < 8; ++kk) {
            const float4 f0 = *(const float4*)(xrow + kk * 32 + 0);
            const float4 f1 = *(const float4*)(xrow + kk * 32 + 4);
            const float4 f2 = *(const float4*)(xrow + kk * 32 + 8);
            const float4 f3 = *(const float4*)(xrow + kk * 32 + 12);
            v16bf bm;
            bm[0]  = (__bf16)f0.x; bm[1]  = (__bf16)f0.y; bm[2]  = (__bf16)f0.z; bm[3]  = (__bf16)f0.w;
            bm[4]  = (__bf16)f1.x; bm[5]  = (__bf16)f1.y; bm[6]  = (__bf16)f1.z; bm[7]  = (__bf16)f1.w;
            bm[8]  = (__bf16)f2.x; bm[9]  = (__bf16)f2.y; bm[10] = (__bf16)f2.z; bm[11] = (__bf16)f2.w;
            bm[12] = (__bf16)f3.x; bm[13] = (__bf16)f3.y; bm[14] = (__bf16)f3.z; bm[15] = (__bf16)f3.w;
            acc = __builtin_amdgcn_wmma_f32_16x16x32_bf16(false, a1[kk], false, bm,
                                                          (short)0, acc, false, false);
        }
        #pragma unroll
        for (int r = 0; r < 8; ++r)
            s_scores[(r + hi * 8) * ROWP + s0 + lo] = acc[r];
    }
    __syncthreads();

    // ---------------- Phase 2: softmax over S per class row ----------------
    // 8 waves x 2 rows; wave32 shuffle reductions. Writes f32 attention to HBM once,
    // and compacts the LDS row to bf16 in place (feeds GEMM 2 A with no cvt).
    #pragma unroll
    for (int rr = 0; rr < 2; ++rr) {
        const int r = wave * 2 + rr;
        float* __restrict__ row  = &s_scores[r * ROWP];
        __bf16* __restrict__ rowb = (__bf16*)row;

        float m = -3.0e38f;
        for (int j = lane; j < S_; j += 32) m = fmaxf(m, row[j]);
        #pragma unroll
        for (int off = 16; off > 0; off >>= 1) m = fmaxf(m, __shfl_xor(m, off, 32));

        float sum = 0.0f;
        for (int j = lane; j < S_; j += 32) sum += __expf(row[j] - m);
        #pragma unroll
        for (int off = 16; off > 0; off >>= 1) sum += __shfl_xor(sum, off, 32);
        const float inv = 1.0f / sum;

        const int c = c0 + r;
        float* __restrict__ ga = (c < C_) ? (attn + ((size_t)b * C_ + c) * S_) : nullptr;
        for (int j = lane; j < S_; j += 32) {
            const float a = __expf(row[j] - m) * inv;
            if (ga) ga[j] = a;          // coalesced HBM write, exactly once
            rowb[j] = (__bf16)a;        // in-place bf16 compaction (read precedes write)
        }
    }

    // ---------------- Phase 3: logits[16][D] = attn(16xS) * x(SxD) ----------------
    // Double-buffered LDS staging of transposed bf16 x chunks; 1 barrier / chunk.
    auto stage = [&](int ks) {
        const int kb = ks * KC;
        __bf16* __restrict__ xs = s_xs[ks & 1];
        #pragma unroll
        for (int it = 0; it < 2; ++it) {
            const int g  = tid + it * 256;        // 512 (k4,d4) groups
            const int k4 = (g >> 6) * 4;          // 0,4,...,28
            const int d4 = (g & 63) * 4;          // 0,4,...,252
            const float4 r0 = *(const float4*)(xb + (size_t)(kb + k4 + 0) * D_ + d4);
            const float4 r1 = *(const float4*)(xb + (size_t)(kb + k4 + 1) * D_ + d4);
            const float4 r2 = *(const float4*)(xb + (size_t)(kb + k4 + 2) * D_ + d4);
            const float4 r3 = *(const float4*)(xb + (size_t)(kb + k4 + 3) * D_ + d4);
            v4bf c0, c1, c2, c3;
            c0[0] = (__bf16)r0.x; c0[1] = (__bf16)r1.x; c0[2] = (__bf16)r2.x; c0[3] = (__bf16)r3.x;
            c1[0] = (__bf16)r0.y; c1[1] = (__bf16)r1.y; c1[2] = (__bf16)r2.y; c1[3] = (__bf16)r3.y;
            c2[0] = (__bf16)r0.z; c2[1] = (__bf16)r1.z; c2[2] = (__bf16)r2.z; c2[3] = (__bf16)r3.z;
            c3[0] = (__bf16)r0.w; c3[1] = (__bf16)r1.w; c3[2] = (__bf16)r2.w; c3[3] = (__bf16)r3.w;
            *(v4bf*)&xs[(d4 + 0) * KP + k4] = c0;   // xs[d][k] : transposed, k-packed b64
            *(v4bf*)&xs[(d4 + 1) * KP + k4] = c1;
            *(v4bf*)&xs[(d4 + 2) * KP + k4] = c2;
            *(v4bf*)&xs[(d4 + 3) * KP + k4] = c3;
        }
    };

    const int n0a = wave * 32;        // each wave owns 2 of the 16 N-tiles of D
    const int n0b = n0a + 16;
    v8f acc0 = {0.f, 0.f, 0.f, 0.f, 0.f, 0.f, 0.f, 0.f};
    v8f acc1 = {0.f, 0.f, 0.f, 0.f, 0.f, 0.f, 0.f, 0.f};

    stage(0);
    for (int ks = 0; ks < NKS; ++ks) {
        __syncthreads();                        // staged chunk ks (and phase 2) visible
        if (ks + 1 < NKS) stage(ks + 1);        // prefetch next chunk into other buffer
        const int kb = ks * KC;
        const __bf16* __restrict__ xs = s_xs[ks & 1];

        // A: bf16 attention row, two aligned 16B LDS reads, no conversion
        const __bf16* __restrict__ rowb = (const __bf16*)(s_scores + lo * ROWP);
        const v8bf a0 = *(const v8bf*)(rowb + kb + hi * 16);
        const v8bf a1v = *(const v8bf*)(rowb + kb + hi * 16 + 8);
        v16bf am;
        #pragma unroll
        for (int j = 0; j < 8; ++j) { am[j] = a0[j]; am[8 + j] = a1v[j]; }

        // B: element j = x[kb + hi*16 + j][n0 + lo] == xs[(n0+lo)*KP + hi*16 + j]
        const v8bf b0l = *(const v8bf*)&xs[(n0a + lo) * KP + hi * 16];
        const v8bf b0h = *(const v8bf*)&xs[(n0a + lo) * KP + hi * 16 + 8];
        const v8bf b1l = *(const v8bf*)&xs[(n0b + lo) * KP + hi * 16];
        const v8bf b1h = *(const v8bf*)&xs[(n0b + lo) * KP + hi * 16 + 8];
        v16bf bm0, bm1;
        #pragma unroll
        for (int j = 0; j < 8; ++j) {
            bm0[j] = b0l[j]; bm0[8 + j] = b0h[j];
            bm1[j] = b1l[j]; bm1[8 + j] = b1h[j];
        }

        acc0 = __builtin_amdgcn_wmma_f32_16x16x32_bf16(false, am, false, bm0,
                                                       (short)0, acc0, false, false);
        acc1 = __builtin_amdgcn_wmma_f32_16x16x32_bf16(false, am, false, bm1,
                                                       (short)0, acc1, false, false);
    }

    #pragma unroll
    for (int r = 0; r < 8; ++r) {
        const int c = c0 + r + hi * 8;
        if (c < C_) {
            float* __restrict__ gl = logits + ((size_t)b * C_ + c) * D_;
            gl[n0a + lo] = acc0[r];
            gl[n0b + lo] = acc1[r];
        }
    }
}

extern "C" void kernel_launch(void* const* d_in, const int* in_sizes, int n_in,
                              void* d_out, int out_size, void* d_ws, size_t ws_size,
                              hipStream_t stream) {
    (void)in_sizes; (void)n_in; (void)out_size; (void)d_ws; (void)ws_size;
    const float* x = (const float*)d_in[0];   // [B, S, D] f32
    const float* W = (const float*)d_in[1];   // [C, D]    f32
    float* logits = (float*)d_out;                               // [B, C, D]
    float* attn   = (float*)d_out + (size_t)B_ * C_ * D_;        // [B, C, S]

    dim3 grid((C_ + CT - 1) / CT, B_);
    labelwise_attn_fused<<<grid, 256, 0, stream>>>(x, W, logits, attn);
}
